// Sketch_RNN_Complete_VAE_91036126806750
// MI455X (gfx1250) — matrix operand order repaired and checked by reading, not compiled, not run on hardware
//
#include <hip/hip_runtime.h>
#include <hip/hip_bf16.h>

// ---------------------------------------------------------------------------
// Sketch-RNN VAE forward for MI455X (gfx1250, wave32, WMMA bf16).
// ---------------------------------------------------------------------------

#define B_    1024
#define T_    256
#define PT_   5
#define EH    256           // encoder hidden
#define EHA   (EH + 32)     // augmented K (hidden | input | pad)
#define KTE   (EHA / 32)    // 9 k-tiles
#define DH    512           // decoder hidden
#define DHA   (DH + 32)     // augmented K
#define KTD   (DHA / 32)    // 17 k-tiles
#define ZD    128           // latent
#define OUTD  123
#define NWG   (B_ / 16)     // 64 persistent workgroups, 16 batch rows each

typedef __attribute__((ext_vector_type(16))) __bf16 v16bf;
typedef __attribute__((ext_vector_type(8)))  __bf16 v8bf;
typedef __attribute__((ext_vector_type(8)))  float  v8f;
typedef __attribute__((ext_vector_type(4)))  float  v4f;

union V16BH { v16bf v; v8bf h[2]; };
union V8FQ  { v8f v; v4f q[2]; };

static __device__ __forceinline__ float sigm(float x) {
  return 1.0f / (1.0f + __expf(-x));
}

static __device__ __forceinline__ v8f wmma_bf16(v16bf a, v16bf b, v8f c) {
  // D = A(16x32) * B(32x16) + C, f32 accumulate
  return __builtin_amdgcn_wmma_f32_16x16x32_bf16(false, a, false, b,
                                                 (short)0, c, false, false);
}

// A fragment (16x32 bf16) from a row-major [16][ldk] activation buffer.
// Per ISA 7.12.2: lane m=L%16 holds K chunks {k0+d..+7, k0+16+d..+7}, d=8*(L>=16).
static __device__ __forceinline__ v16bf load_a(const __bf16* base, int ldk,
                                               int k0, int lane) {
  const int m = lane & 15;
  const int d = (lane & 16) ? 8 : 0;
  const __bf16* p = base + (size_t)m * ldk + k0 + d;
  V16BH u;
  u.h[0] = *(const v8bf*)(p);
  u.h[1] = *(const v8bf*)(p + 16);
  return u.v;
}

// B fragment (32x16) from weights stored row-major [N][K] (W[n][k]); lane
// n=L%16 holds K chunks {k0+d..+7, k0+16+d..+7}, d=8*(L>=16).
static __device__ __forceinline__ v16bf load_b(const __bf16* W, int ldk,
                                               int n0, int k0, int lane) {
  const int n = n0 + (lane & 15);
  const int kb = k0 + ((lane & 16) ? 8 : 0);
  const __bf16* p = W + (size_t)n * ldk + kb;
  V16BH u;
  u.h[0] = *(const v8bf*)(p);
  u.h[1] = *(const v8bf*)(p + 16);
  return u.v;
}

// ---------------------------------------------------------------------------
// Weight convert f32 -> bf16 into a column block of dst with zero padding.
// dst[r*dstLd + dstOff + c] = (r<Rcopy && c<Ccopy) ? src[r*srcLd+srcOff+c] : 0
// ---------------------------------------------------------------------------
__global__ __launch_bounds__(256) void cvt_kernel(
    __bf16* __restrict__ dst, int dstLd, int dstOff,
    const float* __restrict__ src, int R, int Cblk, int Rcopy, int Ccopy,
    int srcLd, int srcOff) {
  const int n = R * Cblk;
  for (int idx = blockIdx.x * 256 + threadIdx.x; idx < n;
       idx += gridDim.x * 256) {
    const int r = idx / Cblk, c = idx % Cblk;
    float v =
        (r < Rcopy && c < Ccopy) ? src[(size_t)r * srcLd + srcOff + c] : 0.f;
    dst[(size_t)r * dstLd + dstOff + c] = (__bf16)v;
  }
}

// ---------------------------------------------------------------------------
// Encoder forward LSTM (persistent). 64 WGs x 16 batch rows; 8 waves; wave w
// owns hidden slice [32w, 32w+32). Cell state in accumulator-layout regs;
// h double-buffered bf16 in LDS, AUGMENTED with the 5-wide input in cols
// EH..EH+4 so input folds into the recurrent GEMM (Waug = [Whh | Wih | 0]).
// K loop is software-pipelined: next A (DS) + 4 B (global) prefetched while
// the current 4 gate WMMAs execute.
// ---------------------------------------------------------------------------
__global__ __launch_bounds__(256) void enc_fwd_kernel(
    const float* __restrict__ s, const __bf16* __restrict__ Waug,
    const float* __restrict__ bias, __bf16* __restrict__ hlast) {
  __shared__ __bf16 hbuf[2][16][EHA];
  const int tid = threadIdx.x;
  const int lane = tid & 31;
  const int w = tid >> 5;
  const int b0 = blockIdx.x * 16;
  const int ln = lane & 15;
  const int dlt = (lane & 16) ? 8 : 0;

  {
    __bf16* hb = &hbuf[0][0][0];
    for (int i = tid; i < 2 * 16 * EHA; i += 256) hb[i] = (__bf16)0.0f;
  }
  __syncthreads();
  if (tid < 16) {  // x_0 = s[:, 0]
    const float* xp = s + (size_t)(b0 + tid) * T_ * PT_;
#pragma unroll
    for (int k = 0; k < PT_; ++k) hbuf[0][tid][EH + k] = (__bf16)xp[k];
  }
  v8f zero8 = {};
  v8f c_reg[2];
  c_reg[0] = zero8;
  c_reg[1] = zero8;
  __syncthreads();

  int cur = 0;
  for (int t = 0; t < T_; ++t) {
    const int nxt = cur ^ 1;
#pragma unroll
    for (int jt = 0; jt < 2; ++jt) {
      const int jbase = (w * 2 + jt) * 16;  // gate g at rows g*EH + jbase
      v8f acc[4];
#pragma unroll
      for (int g = 0; g < 4; ++g) acc[g] = zero8;
      v16bf acur = load_a(&hbuf[cur][0][0], EHA, 0, lane);
      v16bf bcur[4];
#pragma unroll
      for (int g = 0; g < 4; ++g)
        bcur[g] = load_b(Waug, EHA, g * EH + jbase, 0, lane);
#pragma unroll 2
      for (int kt = 0; kt < KTE - 1; ++kt) {
        const v16bf anx = load_a(&hbuf[cur][0][0], EHA, (kt + 1) * 32, lane);
        v16bf bnx[4];
#pragma unroll
        for (int g = 0; g < 4; ++g)
          bnx[g] = load_b(Waug, EHA, g * EH + jbase, (kt + 1) * 32, lane);
#pragma unroll
        for (int g = 0; g < 4; ++g) acc[g] = wmma_bf16(acur, bcur[g], acc[g]);
        acur = anx;
#pragma unroll
        for (int g = 0; g < 4; ++g) bcur[g] = bnx[g];
      }
#pragma unroll
      for (int g = 0; g < 4; ++g) acc[g] = wmma_bf16(acur, bcur[g], acc[g]);

      const int j = jbase + ln;
#pragma unroll
      for (int r = 0; r < 8; ++r) {
        const float ii = sigm(acc[0][r] + bias[0 * EH + j]);
        const float ff = sigm(acc[1][r] + bias[1 * EH + j]);
        const float gg = tanhf(acc[2][r] + bias[2 * EH + j]);
        const float oo = sigm(acc[3][r] + bias[3 * EH + j]);
        const float cc = ff * c_reg[jt][r] + ii * gg;
        c_reg[jt][r] = cc;
        const float hh = oo * tanhf(cc);
        hbuf[nxt][r + dlt][j] = (__bf16)hh;
        if (t == T_ - 1)
          hlast[(size_t)(b0 + r + dlt) * (2 * EH) + j] = (__bf16)hh;
      }
    }
    if (t + 1 < T_ && tid < 16) {  // stage x_{t+1} = s[:, t+1]
      const float* xp = s + ((size_t)(b0 + tid) * T_ + (t + 1)) * PT_;
#pragma unroll
      for (int k = 0; k < PT_; ++k) hbuf[nxt][tid][EH + k] = (__bf16)xp[k];
    }
    cur = nxt;
    __syncthreads();
  }
}

// ---------------------------------------------------------------------------
// Encoder backward direction: the reference only consumes hs_b[-1], which is
// the backward LSTM's FIRST step on x[T-1] from zero state. One tiny step.
// ---------------------------------------------------------------------------
__global__ __launch_bounds__(256) void enc_bwd_kernel(
    const float* __restrict__ s, const float* __restrict__ Wih,
    const float* __restrict__ bias, __bf16* __restrict__ hlast) {
  const int idx = blockIdx.x * 256 + threadIdx.x;
  if (idx >= B_ * EH) return;
  const int b = idx / EH, j = idx % EH;
  const float* xp = s + ((size_t)b * T_ + (T_ - 1)) * PT_;
  float g4[4];
#pragma unroll
  for (int g = 0; g < 4; ++g) {
    const int n = g * EH + j;
    float a = bias[n];
#pragma unroll
    for (int k = 0; k < PT_; ++k) a += xp[k] * Wih[(size_t)n * PT_ + k];
    g4[g] = a;
  }
  const float cc = sigm(g4[0]) * tanhf(g4[2]);  // c_prev == 0
  const float hh = sigm(g4[3]) * tanhf(cc);
  hlast[(size_t)b * (2 * EH) + EH + j] = (__bf16)hh;
}

// ---------------------------------------------------------------------------
// VAE: mu/presig from h_last (K=512), z = mu + exp(presig/2)*eps,
// h0 = tanh(z @ W_h0^T + b). Wave w keeps mu tile and sig tile of the SAME
// latent slice so z is a pure in-register elementwise combine.
// ---------------------------------------------------------------------------
__global__ __launch_bounds__(256) void vae_kernel(
    const __bf16* __restrict__ hlast, const __bf16* __restrict__ Wms,
    const float* __restrict__ bmu, const float* __restrict__ bsig,
    const float* __restrict__ eps, const __bf16* __restrict__ Wh0,
    const float* __restrict__ bh0, __bf16* __restrict__ zbf,
    __bf16* __restrict__ h0bf) {
  __shared__ __bf16 zls[16][ZD];
  const int tid = threadIdx.x;
  const int lane = tid & 31;
  const int w = tid >> 5;
  const int b0 = blockIdx.x * 16;
  const int ln = lane & 15;
  const int dlt = (lane & 16) ? 8 : 0;

  v8f zero8 = {};
  v8f am = zero8, asg = zero8;
  const __bf16* abase = hlast + (size_t)b0 * (2 * EH);
#pragma unroll 2
  for (int kt = 0; kt < (2 * EH) / 32; ++kt) {
    const v16bf af = load_a(abase, 2 * EH, kt * 32, lane);
    am = wmma_bf16(af, load_b(Wms, 2 * EH, w * 16, kt * 32, lane), am);
    asg = wmma_bf16(af, load_b(Wms, 2 * EH, ZD + w * 16, kt * 32, lane), asg);
  }
  const int j = w * 16 + ln;
  const float bm = bmu[j], bs = bsig[j];
#pragma unroll
  for (int r = 0; r < 8; ++r) {
    const int b = b0 + r + dlt;
    const float mu = am[r] + bm;
    const float sg = __expf((asg[r] + bs) * 0.5f);
    const float z = mu + sg * eps[(size_t)b * ZD + j];
    const __bf16 zb = (__bf16)z;
    zls[r + dlt][j] = zb;
    zbf[(size_t)b * ZD + j] = zb;
  }
  __syncthreads();

#pragma unroll 1
  for (int i = 0; i < 4; ++i) {
    const int n0 = (w * 4 + i) * 16;
    v8f a = zero8;
#pragma unroll
    for (int kt = 0; kt < ZD / 32; ++kt)
      a = wmma_bf16(load_a(&zls[0][0], ZD, kt * 32, lane),
                    load_b(Wh0, ZD, n0, kt * 32, lane), a);
    const float bb = bh0[n0 + ln];
#pragma unroll
    for (int r = 0; r < 8; ++r)
      h0bf[(size_t)(b0 + r + dlt) * DH + n0 + ln] = (__bf16)tanhf(a[r] + bb);
  }
}

// ---------------------------------------------------------------------------
// z-part of decoder gates (time-constant): gzT[n][b] = dec_b[n] + z[b]·Wz[n].
// Stored TRANSPOSED [2048][B] so the decoder's accumulator init is two
// coalesced float4 loads per D-fragment.
// ---------------------------------------------------------------------------
__global__ __launch_bounds__(256) void zpart_kernel(
    const __bf16* __restrict__ zbf, const __bf16* __restrict__ Wz,
    const float* __restrict__ decb, float* __restrict__ gzT) {
  const int tid = threadIdx.x;
  const int lane = tid & 31;
  const int w = tid >> 5;
  const int b0 = blockIdx.x * 16;
  const int ln = lane & 15;
  const int dlt = (lane & 16) ? 8 : 0;

  const __bf16* abase = zbf + (size_t)b0 * ZD;
  v8f zero8 = {};
#pragma unroll 1
  for (int i = 0; i < 16; ++i) {
    const int n0 = (w * 16 + i) * 16;
    v8f a = zero8;
#pragma unroll
    for (int kt = 0; kt < ZD / 32; ++kt)
      a = wmma_bf16(load_a(abase, ZD, kt * 32, lane),
                    load_b(Wz, ZD, n0, kt * 32, lane), a);
    const int n = n0 + ln;
    const float bb = decb[n];
    V8FQ u;
#pragma unroll
    for (int r = 0; r < 8; ++r) u.v[r] = a[r] + bb;
    v4f* gp = (v4f*)(gzT + (size_t)n * B_ + b0 + dlt);
    gp[0] = u.q[0];
    gp[1] = u.q[1];
  }
}

// ---------------------------------------------------------------------------
// Decoder LSTM + GMM output head (persistent). 64 WGs x 16 batch rows; wave w
// owns hidden slice [64w, 64w+64). c in registers; h double-buffered bf16 in
// LDS augmented with the 5-wide teacher-forced input (Waug = [Whh | Wih_s | 0],
// 2.2MB bf16 streamed from L2 each step). z-part+bias accumulator init uses
// VOLATILE loads so LICM cannot hoist 128 VGPRs of it out of the t-loop.
// Software-pipelined K loop; output transforms parallelized 16-wide per row.
// ---------------------------------------------------------------------------
__global__ __launch_bounds__(256) void dec_kernel(
    const float* __restrict__ s, const __bf16* __restrict__ h0bf,
    const float* __restrict__ gzT, const __bf16* __restrict__ Waug,
    const __bf16* __restrict__ Wout, const float* __restrict__ bout,
    float* __restrict__ out, float* __restrict__ qkpart) {
  __shared__ __bf16 hbuf[2][16][DHA];
  __shared__ float yls[16][128];
  __shared__ float qs[16];
  const int tid = threadIdx.x;
  const int lane = tid & 31;
  const int w = tid >> 5;
  const int b0 = blockIdx.x * 16;
  const int ln = lane & 15;
  const int dlt = (lane & 16) ? 8 : 0;

  {
    __bf16* hb = &hbuf[0][0][0];
    for (int i = tid; i < 2 * 16 * DHA; i += 256) hb[i] = (__bf16)0.0f;
  }
  __syncthreads();
  for (int i = tid; i < 16 * DH; i += 256)
    hbuf[0][i / DH][i % DH] = h0bf[(size_t)(b0 + i / DH) * DH + i % DH];
  if (tid < 16) hbuf[0][tid][DH + 2] = (__bf16)1.0f;  // s0 = (0,0,1,0,0)

  v8f zero8 = {};
  v8f c_reg[4];
#pragma unroll
  for (int jt = 0; jt < 4; ++jt) c_reg[jt] = zero8;
  __syncthreads();

  int cur = 0;
  for (int t = 0; t < T_; ++t) {
    const int nxt = cur ^ 1;
#pragma unroll
    for (int jt = 0; jt < 4; ++jt) {
      const int jbase = (w * 4 + jt) * 16;  // gate g at rows g*DH + jbase
      v8f acc[4];
#pragma unroll
      for (int g = 0; g < 4; ++g) {
        // acc init = z-part + bias; volatile blocks LICM register-caching.
        const volatile v4f* gp =
            (const volatile v4f*)(gzT + (size_t)(g * DH + jbase + ln) * B_ +
                                  b0 + dlt);
        V8FQ ug;
        ug.q[0] = gp[0];
        ug.q[1] = gp[1];
        acc[g] = ug.v;
      }
      v16bf acur = load_a(&hbuf[cur][0][0], DHA, 0, lane);
      v16bf bcur[4];
#pragma unroll
      for (int g = 0; g < 4; ++g)
        bcur[g] = load_b(Waug, DHA, g * DH + jbase, 0, lane);
#pragma unroll 2
      for (int kt = 0; kt < KTD - 1; ++kt) {
        const v16bf anx = load_a(&hbuf[cur][0][0], DHA, (kt + 1) * 32, lane);
        v16bf bnx[4];
#pragma unroll
        for (int g = 0; g < 4; ++g)
          bnx[g] = load_b(Waug, DHA, g * DH + jbase, (kt + 1) * 32, lane);
#pragma unroll
        for (int g = 0; g < 4; ++g) acc[g] = wmma_bf16(acur, bcur[g], acc[g]);
        acur = anx;
#pragma unroll
        for (int g = 0; g < 4; ++g) bcur[g] = bnx[g];
      }
#pragma unroll
      for (int g = 0; g < 4; ++g) acc[g] = wmma_bf16(acur, bcur[g], acc[g]);

      const int j = jbase + ln;
#pragma unroll
      for (int r = 0; r < 8; ++r) {
        const float ii = sigm(acc[0][r]);
        const float ff = sigm(acc[1][r]);
        const float gg = tanhf(acc[2][r]);
        const float oo = sigm(acc[3][r]);
        const float cc = ff * c_reg[jt][r] + ii * gg;
        c_reg[jt][r] = cc;
        hbuf[nxt][r + dlt][j] = (__bf16)(oo * tanhf(cc));
      }
    }
    if (t + 1 < T_ && tid < 16) {  // dec_in[t+1] = s[:, t]
      const float* xp = s + ((size_t)(b0 + tid) * T_ + t) * PT_;
#pragma unroll
      for (int k = 0; k < PT_; ++k) hbuf[nxt][tid][DH + k] = (__bf16)xp[k];
    }
    __syncthreads();  // new h complete

    // y = h_new @ W_out^T (padded to N=128); wave w -> n-tile w. Pipelined.
    {
      v8f a = zero8;
      const int n0 = w * 16;
      v16bf acur = load_a(&hbuf[nxt][0][0], DHA, 0, lane);
      v16bf bcur = load_b(Wout, DH, n0, 0, lane);
#pragma unroll 2
      for (int kt = 0; kt < DH / 32 - 1; ++kt) {
        const v16bf anx = load_a(&hbuf[nxt][0][0], DHA, (kt + 1) * 32, lane);
        const v16bf bnx = load_b(Wout, DH, n0, (kt + 1) * 32, lane);
        a = wmma_bf16(acur, bcur, a);
        acur = anx;
        bcur = bnx;
      }
      a = wmma_bf16(acur, bcur, a);
#pragma unroll
      for (int r = 0; r < 8; ++r) yls[r + dlt][n0 + ln] = a[r];
    }
    __syncthreads();

    // Output transforms: 16 threads per batch row (row = tid>>4, sub = tid&15).
    // Each sub redundantly computes the 20-logit softmax denominator (LDS
    // broadcast reads), then writes outputs o = sub, sub+16, ... of 123.
    {
      const int row = tid >> 4;
      const int sub = tid & 15;
      float* orow = out + ((size_t)t * B_ + b0 + row) * OUTD;
      float mx = -3.0e38f;
#pragma unroll
      for (int q = 0; q < 20; ++q)
        mx = fmaxf(mx, yls[row][q * 6] + bout[q * 6]);
      float den = 0.f;
#pragma unroll
      for (int q = 0; q < 20; ++q)
        den += __expf(yls[row][q * 6] + bout[q * 6] - mx);
      const float inv = 1.f / den;
#pragma unroll
      for (int it = 0; it < 8; ++it) {
        const int o = sub + it * 16;
        if (o < OUTD) {
          float v;
          if (o < 20) {
            v = __expf(yls[row][o * 6] + bout[o * 6] - mx) * inv;
          } else if (o < 40) {
            const int q = o - 20;
            v = yls[row][q * 6 + 1] + bout[q * 6 + 1];
          } else if (o < 60) {
            const int q = o - 40;
            v = yls[row][q * 6 + 2] + bout[q * 6 + 2];
          } else if (o < 80) {
            const int q = o - 60;
            v = __expf(yls[row][q * 6 + 3] + bout[q * 6 + 3]);
          } else if (o < 100) {
            const int q = o - 80;
            v = __expf(yls[row][q * 6 + 4] + bout[q * 6 + 4]);
          } else if (o < 120) {
            const int q = o - 100;
            v = tanhf(yls[row][q * 6 + 5] + bout[q * 6 + 5]);
          } else {
            v = __expf(yls[row][o] + bout[o]);  // raw exp_qk (normalized later)
          }
          orow[o] = v;
        }
      }
      if (sub == 0) {
        const float e0 = __expf(yls[row][120] + bout[120]);
        const float e1 = __expf(yls[row][121] + bout[121]);
        const float e2 = __expf(yls[row][122] + bout[122]);
        qs[row] = e0 + e1 + e2;
      }
    }
    __syncthreads();
    if (tid == 0) {
      float ssum = 0.f;
#pragma unroll
      for (int i = 0; i < 16; ++i) ssum += qs[i];
      qkpart[(size_t)t * gridDim.x + blockIdx.x] = ssum;
    }
    cur = nxt;
    __syncthreads();
  }
}

// ---------------------------------------------------------------------------
// qk normalization by the GLOBAL (per-timestep) sum, deterministic reduction
// of the 64 per-WG partials.
// ---------------------------------------------------------------------------
__global__ __launch_bounds__(256) void qknorm_kernel(
    float* __restrict__ out, const float* __restrict__ qkpart, int nwg) {
  const int idx = blockIdx.x * 256 + threadIdx.x;
  if (idx >= T_ * B_ * 3) return;
  const int t = idx / (B_ * 3);
  const int r = idx % (B_ * 3);
  const int b = r / 3, k = r % 3;
  float den = 0.f;
  for (int i = 0; i < nwg; ++i) den += qkpart[(size_t)t * nwg + i];
  out[((size_t)t * B_ + b) * OUTD + 120 + k] /= den;
}

// ---------------------------------------------------------------------------
extern "C" void kernel_launch(void* const* d_in, const int* in_sizes, int n_in,
                              void* d_out, int out_size, void* d_ws,
                              size_t ws_size, hipStream_t stream) {
  (void)in_sizes;
  (void)n_in;
  (void)out_size;
  (void)ws_size;
  const float* s = (const float*)d_in[0];
  const float* eps = (const float*)d_in[1];
  const float* encWihF = (const float*)d_in[2];
  const float* encWhhF = (const float*)d_in[3];
  const float* encBF = (const float*)d_in[4];
  const float* encWihB = (const float*)d_in[5];
  const float* encBB = (const float*)d_in[7];
  const float* Wmu = (const float*)d_in[8];
  const float* bmu = (const float*)d_in[9];
  const float* Wsig = (const float*)d_in[10];
  const float* bsig = (const float*)d_in[11];
  const float* Wh0 = (const float*)d_in[12];
  const float* bh0 = (const float*)d_in[13];
  const float* decWih = (const float*)d_in[14];
  const float* decWhh = (const float*)d_in[15];
  const float* decB = (const float*)d_in[16];
  const float* Wout = (const float*)d_in[17];
  const float* bout = (const float*)d_in[18];

  char* wsb = (char*)d_ws;
  size_t off = 0;
  auto take = [&](size_t bytes) -> void* {
    void* p = (void*)(wsb + off);
    off = (off + bytes + 255) & ~(size_t)255;
    return p;
  };

  __bf16* WaugE = (__bf16*)take((size_t)1024 * EHA * 2);   // [Whh | Wih | 0]
  __bf16* Wms = (__bf16*)take((size_t)256 * 512 * 2);
  __bf16* Wh0bf = (__bf16*)take((size_t)512 * 128 * 2);
  __bf16* WaugD = (__bf16*)take((size_t)2048 * DHA * 2);   // [Whh | Wih_s | 0]
  __bf16* Wzbf = (__bf16*)take((size_t)2048 * 128 * 2);
  __bf16* WoutBf = (__bf16*)take((size_t)128 * 512 * 2);
  __bf16* hlastBf = (__bf16*)take((size_t)1024 * 512 * 2);
  __bf16* zbfp = (__bf16*)take((size_t)1024 * 128 * 2);
  __bf16* h0bfp = (__bf16*)take((size_t)1024 * 512 * 2);
  float* gzT = (float*)take((size_t)2048 * 1024 * 4);
  float* qkpart = (float*)take((size_t)T_ * NWG * 4);

  auto cvt = [&](__bf16* dst, int dstLd, int dstOff, const float* src, int R,
                 int Cblk, int Rcopy, int Ccopy, int srcLd, int srcOff) {
    const int n = R * Cblk;
    cvt_kernel<<<(n + 255) / 256, 256, 0, stream>>>(
        dst, dstLd, dstOff, src, R, Cblk, Rcopy, Ccopy, srcLd, srcOff);
  };

  // Encoder augmented weights: cols 0..255 = Whh, cols 256..260 = Wih, pad 0.
  cvt(WaugE, EHA, 0, encWhhF, 1024, EH, 1024, EH, EH, 0);
  cvt(WaugE, EHA, EH, encWihF, 1024, 32, 1024, PT_, PT_, 0);
  // VAE weights: rows 0..127 = W_mu, rows 128..255 = W_sig.
  cvt(Wms, 512, 0, Wmu, 128, 512, 128, 512, 512, 0);
  cvt(Wms + (size_t)128 * 512, 512, 0, Wsig, 128, 512, 128, 512, 512, 0);
  cvt(Wh0bf, 128, 0, Wh0, 512, 128, 512, 128, 128, 0);
  // Decoder augmented weights: cols 0..511 = Whh, cols 512..516 = Wih_s.
  cvt(WaugD, DHA, 0, decWhh, 2048, DH, 2048, DH, DH, 0);
  cvt(WaugD, DHA, DH, decWih, 2048, 32, 2048, PT_, 133, 0);
  cvt(Wzbf, 128, 0, decWih, 2048, 128, 2048, 128, 133, PT_);
  cvt(WoutBf, 512, 0, Wout, 128, 512, 123, 512, 512, 0);  // pad N 123->128

  enc_fwd_kernel<<<NWG, 256, 0, stream>>>(s, WaugE, encBF, hlastBf);
  enc_bwd_kernel<<<(B_ * EH + 255) / 256, 256, 0, stream>>>(s, encWihB, encBB,
                                                            hlastBf);
  vae_kernel<<<NWG, 256, 0, stream>>>(hlastBf, Wms, bmu, bsig, eps, Wh0bf, bh0,
                                      zbfp, h0bfp);
  zpart_kernel<<<NWG, 256, 0, stream>>>(zbfp, Wzbf, decB, gzT);
  dec_kernel<<<NWG, 256, 0, stream>>>(s, h0bfp, gzT, WaugD, WoutBf, bout,
                                      (float*)d_out, qkpart);
  const int qn = T_ * B_ * 3;
  qknorm_kernel<<<(qn + 255) / 256, 256, 0, stream>>>((float*)d_out, qkpart,
                                                      NWG);
}
